// TGAN_69226282876948
// MI455X (gfx1250) — matrix operand order repaired and checked by reading, not compiled, not run on hardware
//
#include <hip/hip_runtime.h>
#include <stdint.h>

#define DEVINL __device__ __forceinline__

typedef __attribute__((ext_vector_type(16))) __bf16 v16bf;
typedef __attribute__((ext_vector_type(8)))  float  v8f;

#if __has_builtin(__builtin_amdgcn_tensor_load_to_lds)
#define HAVE_TDM 1
typedef __attribute__((ext_vector_type(4))) unsigned int tdm_u4;
typedef __attribute__((ext_vector_type(8))) int          tdm_i8;
typedef __attribute__((ext_vector_type(4))) int          tdm_i4;
#else
#define HAVE_TDM 0
#endif

// ---------------- problem sizes ----------------
constexpr int Bsz = 4096, Nseq = 32, Hh = 4, Mm = 768, DKk = 192;
constexpr int ROWS_KV = Bsz * Nseq;             // 131072

// ---------------- workspace layout (bytes) ----------------
constexpr size_t SZ_W768 = (size_t)768 * 768 * 2;
constexpr size_t OFF_WQ = 0;
constexpr size_t OFF_WK = OFF_WQ + SZ_W768;
constexpr size_t OFF_WV = OFF_WK + SZ_W768;
constexpr size_t OFF_WO = OFF_WV + SZ_W768;
constexpr size_t OFF_W1 = OFF_WO + SZ_W768;                 // 1024x256 bf16
constexpr size_t OFF_W2 = OFF_W1 + (size_t)1024 * 256 * 2;  // 256x256 bf16
constexpr size_t OFF_Q  = OFF_W2 + (size_t)256 * 256 * 2;   // 4096x768 bf16
constexpr size_t OFF_K  = OFF_Q  + (size_t)Bsz * Mm * 2;    // 131072x768 bf16
constexpr size_t OFF_V  = OFF_K  + (size_t)ROWS_KV * Mm * 2;

// ---------------- helpers ----------------
DEVINL unsigned short f2bfu(float f) {
  unsigned u = __float_as_uint(f);
  return (unsigned short)((u + 0x7fffu + ((u >> 16) & 1u)) >> 16);   // RNE
}
DEVINL unsigned packbf(float a, float b) {
  return (unsigned)f2bfu(a) | ((unsigned)f2bfu(b) << 16);
}
DEVINL float bflo2f(unsigned u) { return __uint_as_float(u << 16); }
DEVINL float bfhi2f(unsigned u) { return __uint_as_float(u & 0xffff0000u); }
DEVINL float bfs2f(unsigned short v) { return __uint_as_float((unsigned)v << 16); }

DEVINL v8f v8zero() {
  v8f z;
#pragma unroll
  for (int i = 0; i < 8; ++i) z[i] = 0.f;
  return z;
}

DEVINL v8f wmma_bf16(v16bf a, v16bf b, v8f c) {
  // (neg_a, A, neg_b, B, c_mod, C, reuse_a, reuse_b)
  return __builtin_amdgcn_wmma_f32_16x16x32_bf16(false, a, false, b, (short)0, c,
                                                 false, false);
}

// B fragment (32x16 bf16) pre-swizzled in global ws: 1024 B per fragment,
// lane-major: lane l owns 32 contiguous bytes -> two b128 loads, fully coalesced.
DEVINL v16bf loadB(const unsigned short* base, int frag, int lane) {
  const uint4* p = (const uint4*)(base + (size_t)frag * 512) + lane * 2;
  union { v16bf v; uint4 q[2]; } t;
  t.q[0] = p[0];
  t.q[1] = p[1];
  return t.v;
}

// A fragment (16x32 bf16) gathered from row-major bf16 LDS, stride in halfwords
// (must be even). Per ISA 7.12.2: lane = m + 16*hi ; VGPR j<4: K = hi*8+2j,
// VGPR j>=4: K = 16 + hi*8 + 2(j-4).
DEVINL v16bf loadA_lds(const unsigned short* s, int strideH, int row0, int k0,
                       int lane) {
  const unsigned* b = (const unsigned*)s;
  int m = row0 + (lane & 15);
  int hi = lane >> 4;
  int base = m * strideH + k0 + hi * 8;
  union { v16bf v; unsigned u[8]; } t;
#pragma unroll
  for (int j = 0; j < 4; ++j) t.u[j] = b[(base + 2 * j) >> 1];
#pragma unroll
  for (int j = 0; j < 4; ++j) t.u[4 + j] = b[(base + 16 + 2 * j) >> 1];
  return t.v;
}

#if HAVE_TDM
// 1-D TDM copy: n2B bf16 elements, global -> LDS (compact), one issue per call.
// D# per CDNA5 ISA: group0 = {count/lds/global/type},
// group1 = {mask=0, data_size=2B, dims}, groups 2/3 unused (zero).
DEVINL void tdm_load_1d(void* lds_dst, const void* gsrc, unsigned n2B) {
  unsigned lds = (unsigned)(uintptr_t)lds_dst;          // low 32b = LDS offset
  unsigned long long ga = (unsigned long long)(uintptr_t)gsrc;
  tdm_u4 g0;
  g0[0] = 1u;                                           // count=1, user D#
  g0[1] = lds;                                          // lds_addr
  g0[2] = (unsigned)(ga & 0xffffffffu);                 // global_addr[31:0]
  g0[3] = (unsigned)((ga >> 32) & 0x01ffffffu) | (2u << 30);  // addr[56:32]|type=2
  tdm_i8 g1;
  g1[0] = 0x00010000;                   // workgroup_mask=0, data_size=1 (2 bytes)
  g1[1] = (int)((n2B & 0xffffu) << 16); // tensor_dim0[15:0] @ bits 63:48
  g1[2] = (int)(((n2B >> 16) & 0xffffu) | (1u << 16)); // dim0[31:16], tensor_dim1=1
  g1[3] = (int)((n2B & 0xffffu) << 16); // tile_dim0 @ bits 127:112
  g1[4] = 1;                            // tile_dim1=1, tile_dim2=0
  g1[5] = (int)n2B;                     // tensor_dim0_stride[31:0]
  g1[6] = 0;
  g1[7] = 0;
  tdm_i4 g2 = (tdm_i4)0;
  tdm_i4 g3 = (tdm_i4)0;
#if defined(__clang_major__) && (__clang_major__ >= 23)
  tdm_i8 g4 = (tdm_i8)0;
  __builtin_amdgcn_tensor_load_to_lds(g0, g1, g2, g3, g4, 0);
#else
  __builtin_amdgcn_tensor_load_to_lds(g0, g1, g2, g3, 0);
#endif
}
#endif

// ---------------- kernel: weight f32 -> bf16 B-fragment swizzle ----------------
__global__ void prep_weight(const float* __restrict__ W, unsigned* __restrict__ o,
                            int Kdim, int Ndim) {
  int p = blockIdx.x * blockDim.x + threadIdx.x;
  int total = (Kdim >> 1) * Ndim;
  if (p >= total) return;
  int n = p % Ndim;
  int k0 = (p / Ndim) << 1;                 // even k; pair (k0, k0+1) -> one dword
  float f0 = W[(size_t)k0 * Ndim + n];
  float f1 = W[(size_t)(k0 + 1) * Ndim + n];
  int kc = k0 >> 5, kl = k0 & 31;
  int nt = n >> 4, nn = n & 15;
  int lane = nn + (((kl >> 3) & 1) << 4);   // lanes 16..31 hold K=8..15,24..31
  int j = ((kl >> 4) << 2) | ((kl & 7) >> 1); // dword within lane
  int frag = kc * (Ndim >> 4) + nt;
  o[(size_t)frag * 256 + lane * 8 + j] = packbf(f0, f1);
}

// ---------------- kernel: K & V projections (A shared) ----------------
// A block: 128 rows of k_in = [seq | seq_e | seq_t] (bf16 in LDS, stride 776)
__global__ __launch_bounds__(256) void kv_gemm(
    const float* __restrict__ seq, const float* __restrict__ seq_e,
    const float* __restrict__ seq_t, const unsigned short* __restrict__ WkB,
    const unsigned short* __restrict__ WvB, unsigned short* __restrict__ Kmat,
    unsigned short* __restrict__ Vmat) {
  extern __shared__ unsigned short sA[];    // 128 x 776 bf16
  constexpr int SAst = 776;
  const int tid = threadIdx.x;
  const int row0 = blockIdx.x * 128;
  unsigned* sAu = (unsigned*)sA;
  for (int p = tid; p < 128 * 384; p += 256) {
    int r = p / 384, cp = p % 384, c0 = cp * 2;
    size_t rb = (size_t)(row0 + r) * 256;
    float2 f;
    if (c0 < 256)      f = *(const float2*)(seq + rb + c0);
    else if (c0 < 512) f = *(const float2*)(seq_e + rb + (c0 - 256));
    else               f = *(const float2*)(seq_t + rb + (c0 - 512));
    sAu[r * (SAst / 2) + cp] = packbf(f.x, f.y);
  }
  __syncthreads();

  const int w = tid >> 5, lane = tid & 31;
  for (int ni = 0; ni < 6; ++ni) {
    int nt = w + ni * 8;                     // 0..47
    v8f aK[8], aV[8];
#pragma unroll
    for (int mt = 0; mt < 8; ++mt) { aK[mt] = v8zero(); aV[mt] = v8zero(); }
    for (int kc = 0; kc < 24; ++kc) {
      if (kc + 1 < 24) {
        __builtin_prefetch(WkB + (size_t)((kc + 1) * 48 + nt) * 512 + lane * 16, 0, 1);
        __builtin_prefetch(WvB + (size_t)((kc + 1) * 48 + nt) * 512 + lane * 16, 0, 1);
      }
      v16bf bk = loadB(WkB, kc * 48 + nt, lane);
      v16bf bv = loadB(WvB, kc * 48 + nt, lane);
      // preload ALL 8 A fragments first so the 16 WMMAs below issue
      // back-to-back with no WAR hazard against the ds loads
      v16bf af[8];
#pragma unroll
      for (int mt = 0; mt < 8; ++mt)
        af[mt] = loadA_lds(sA, SAst, mt * 16, kc * 32, lane);
#pragma unroll
      for (int mt = 0; mt < 8; ++mt) {
        aK[mt] = wmma_bf16(af[mt], bk, aK[mt]);
        aV[mt] = wmma_bf16(af[mt], bv, aV[mt]);
      }
    }
    int n = nt * 16 + (lane & 15);
    int rb = 8 * (lane >> 4);                // C: row = r + 8*hi
#pragma unroll
    for (int mt = 0; mt < 8; ++mt) {
#pragma unroll
      for (int r = 0; r < 8; ++r) {
        size_t m = (size_t)(row0 + mt * 16 + rb + r);
        Kmat[m * 768 + n] = f2bfu(aK[mt][r]);
        Vmat[m * 768 + n] = f2bfu(aV[mt][r]);
      }
    }
  }
}

// ---------------- kernel: Q projection ----------------
// q_in = [src | 0 | src_t], rows = batch index
__global__ __launch_bounds__(256) void q_gemm(
    const float* __restrict__ src, const float* __restrict__ src_t,
    const unsigned short* __restrict__ WqB, unsigned short* __restrict__ Qmat) {
  extern __shared__ unsigned short sA[];    // 128 x 776 bf16
  constexpr int SAst = 776;
  const int tid = threadIdx.x;
  const int row0 = blockIdx.x * 128;
  unsigned* sAu = (unsigned*)sA;
  for (int p = tid; p < 128 * 384; p += 256) {
    int r = p / 384, cp = p % 384, c0 = cp * 2;
    size_t rb = (size_t)(row0 + r) * 256;
    unsigned pk;
    if (c0 < 256) { float2 f = *(const float2*)(src + rb + c0); pk = packbf(f.x, f.y); }
    else if (c0 < 512) pk = 0u;
    else { float2 f = *(const float2*)(src_t + rb + (c0 - 512)); pk = packbf(f.x, f.y); }
    sAu[r * (SAst / 2) + cp] = pk;
  }
  __syncthreads();

  const int w = tid >> 5, lane = tid & 31;
  for (int ni = 0; ni < 6; ++ni) {
    int nt = w + ni * 8;
    v8f acc[8];
#pragma unroll
    for (int mt = 0; mt < 8; ++mt) acc[mt] = v8zero();
    for (int kc = 0; kc < 24; ++kc) {
      v16bf bq = loadB(WqB, kc * 48 + nt, lane);
      v16bf af[8];
#pragma unroll
      for (int mt = 0; mt < 8; ++mt)
        af[mt] = loadA_lds(sA, SAst, mt * 16, kc * 32, lane);
#pragma unroll
      for (int mt = 0; mt < 8; ++mt)
        acc[mt] = wmma_bf16(af[mt], bq, acc[mt]);
    }
    int n = nt * 16 + (lane & 15);
    int rb = 8 * (lane >> 4);
#pragma unroll
    for (int mt = 0; mt < 8; ++mt)
#pragma unroll
      for (int r = 0; r < 8; ++r)
        Qmat[(size_t)(row0 + mt * 16 + rb + r) * 768 + n] = f2bfu(acc[mt][r]);
  }
}

// ---------------- kernel: attention + Wo + LN + MLP (16 batches / WG) ---------
// sQ/sK/sV are compact (stride 768 halfwords) because the TDM writes tiles
// compacted into LDS; the score loop rotates its K traversal per-lane to avoid
// same-bank row starts (384-dword row stride == bank-aligned).
__global__ __launch_bounds__(256) void attn_fused(
    const unsigned short* __restrict__ Qmat, const unsigned short* __restrict__ Kmat,
    const unsigned short* __restrict__ Vmat, const float* __restrict__ src,
    const float* __restrict__ src_t, const unsigned char* __restrict__ mask,
    const unsigned short* __restrict__ WoB, const unsigned short* __restrict__ W1B,
    const unsigned short* __restrict__ W2B, const float* __restrict__ bo,
    const float* __restrict__ ln_g, const float* __restrict__ ln_b,
    const float* __restrict__ b1, const float* __restrict__ b2,
    float* __restrict__ out) {
  extern __shared__ char smem[];
  unsigned short* sQ  = (unsigned short*)(smem + 0);        // 16x768 bf16
  unsigned short* sK  = (unsigned short*)(smem + 24576);    // 32x768 bf16
  unsigned short* sV  = (unsigned short*)(smem + 73728);    // 32x768 bf16
  unsigned short* sAo = (unsigned short*)(smem + 122880);   // 16x776 bf16
  float*          sH  = (float*)(smem + 147712);            // 16x776 f32
  unsigned short* sX  = (unsigned short*)(smem + 197376);   // 16x1032 bf16
  unsigned short* sXH = (unsigned short*)(smem + 230400);   // 16x264 bf16
  float*          sS  = (float*)(smem + 238848);            // 4x32 attn probs

  const int tid = threadIdx.x;
  const int b0 = blockIdx.x * 16;
  const int w = tid >> 5, lane = tid & 31;
  const float scale = 0.07216878364870323f;                 // 1/sqrt(192)

  // stage projected q rows (contiguous 16x768 bf16)
#if HAVE_TDM
  if (w == 0) tdm_load_1d(sQ, Qmat + (size_t)b0 * 768, 16 * 768);
#else
  {
    unsigned* d = (unsigned*)sQ;
    const unsigned* g = (const unsigned*)(Qmat + (size_t)b0 * 768);
    for (int p = tid; p < 16 * 384; p += 256) d[p] = g[p];
  }
#endif

  for (int bl = 0; bl < 16; ++bl) {
    const int b = b0 + bl;
    __syncthreads();   // prior-iteration consumers done before restaging
#if HAVE_TDM
    if (w == 0) {
      tdm_load_1d(sK, Kmat + (size_t)b * 32 * 768, 32 * 768);
      tdm_load_1d(sV, Vmat + (size_t)b * 32 * 768, 32 * 768);
      __builtin_amdgcn_s_wait_tensorcnt(0);
    }
#else
    {
      unsigned* dk = (unsigned*)sK;
      unsigned* dv = (unsigned*)sV;
      const unsigned* gk = (const unsigned*)(Kmat + (size_t)b * 32 * 768);
      const unsigned* gv = (const unsigned*)(Vmat + (size_t)b * 32 * 768);
      for (int p = tid; p < 32 * 384; p += 256) {
        dk[p] = gk[p];
        dv[p] = gv[p];
      }
    }
#endif
    __syncthreads();

    // scores + softmax: wave = head, lane = key position
    if (w < 4) {
      const int h = w, n = lane;
      const unsigned* qp = (const unsigned*)sQ + (bl * 384 + h * 96);
      const unsigned* kp = (const unsigned*)sK + (n * 384 + h * 96);
      float s = 0.f;
#pragma unroll 4
      for (int t = 0; t < 96; ++t) {
        int d2 = t + n;                 // per-lane rotation: spread LDS banks
        d2 -= (d2 >= 96) ? 96 : 0;
        unsigned qu = qp[d2], ku = kp[d2];
        s += bflo2f(qu) * bflo2f(ku) + bfhi2f(qu) * bfhi2f(ku);
      }
      s *= scale;
      if (mask[b * 32 + n]) s = -1e10f;
      float mx = s;
#pragma unroll
      for (int off = 16; off > 0; off >>= 1) mx = fmaxf(mx, __shfl_xor(mx, off, 32));
      float e = __expf(s - mx);
      float se = e;
#pragma unroll
      for (int off = 16; off > 0; off >>= 1) se += __shfl_xor(se, off, 32);
      float a = e / se;
      sS[h * 32 + n] = a;
      // attn_flat[h*B + b][n], appended after out2 (B*256 floats)
      out[(size_t)Bsz * 256 + ((size_t)h * Bsz + b) * 32 + n] = a;
    }
    __syncthreads();

    // context row: out[m] = sum_n attn[h(m)][n] * V[n][m]
#pragma unroll
    for (int i = 0; i < 3; ++i) {
      int m = tid + 256 * i;
      int h = m / 192;
      const float* ap = sS + h * 32;
      float acc = 0.f;
#pragma unroll 8
      for (int n = 0; n < 32; ++n) acc += ap[n] * bfs2f(sV[n * 768 + m]);
      sAo[bl * 776 + m] = f2bfu(acc);
    }
  }
  __syncthreads();

  // ---- Wo projection: sAo (16x768) @ WoB -> sH (f32) ----
  for (int ni = 0; ni < 6; ++ni) {
    int nt = w + ni * 8;
    v8f acc = v8zero();
    for (int kc = 0; kc < 24; ++kc) {
      v16bf a = loadA_lds(sAo, 776, 0, kc * 32, lane);
      v16bf bb = loadB(WoB, kc * 48 + nt, lane);
      acc = wmma_bf16(a, bb, acc);
    }
    int n = nt * 16 + (lane & 15);
    int rb = 8 * (lane >> 4);
#pragma unroll
    for (int r = 0; r < 8; ++r) sH[(rb + r) * 776 + n] = acc[r];
  }
  __syncthreads();

  // ---- residual + LayerNorm + build MLP input sX = [h | src] (bf16) ----
  {
    const int row = tid >> 4;       // 0..15
    const int l16 = tid & 15;
    const int b = b0 + row;
    float sum = 0.f, sq = 0.f;
    for (int m = l16; m < 768; m += 16) {
      float qin;
      if (m < 256)      qin = src[(size_t)b * 256 + m];
      else if (m < 512) qin = 0.f;
      else              qin = src_t[(size_t)b * 256 + (m - 512)];
      float hv = sH[row * 776 + m] + bo[m] + qin;
      sH[row * 776 + m] = hv;
      sum += hv;
      sq += hv * hv;
    }
#pragma unroll
    for (int off = 8; off > 0; off >>= 1) {
      sum += __shfl_xor(sum, off, 16);
      sq  += __shfl_xor(sq, off, 16);
    }
    float mu = sum * (1.f / 768.f);
    float var = sq * (1.f / 768.f) - mu * mu;
    float rstd = rsqrtf(var + 1e-5f);
    for (int m = l16; m < 768; m += 16) {
      float hn = (sH[row * 776 + m] - mu) * rstd * ln_g[m] + ln_b[m];
      sX[row * 1032 + m] = f2bfu(hn);
    }
    for (int c = l16; c < 256; c += 16)
      sX[row * 1032 + 768 + c] = f2bfu(src[(size_t)b * 256 + c]);
  }
  __syncthreads();

  // ---- MLP layer 1: sX (16x1024) @ W1B, +b1, relu -> sXH (bf16) ----
  for (int ni = 0; ni < 2; ++ni) {
    int nt = w + ni * 8;            // 0..15
    v8f acc = v8zero();
    for (int kc = 0; kc < 32; ++kc) {
      v16bf a = loadA_lds(sX, 1032, 0, kc * 32, lane);
      v16bf bb = loadB(W1B, kc * 16 + nt, lane);
      acc = wmma_bf16(a, bb, acc);
    }
    int n = nt * 16 + (lane & 15);
    int rb = 8 * (lane >> 4);
    float bv = b1[n];
#pragma unroll
    for (int r = 0; r < 8; ++r)
      sXH[(rb + r) * 264 + n] = f2bfu(fmaxf(acc[r] + bv, 0.f));
  }
  __syncthreads();

  // ---- MLP layer 2: sXH (16x256) @ W2B, +b2 -> out2 ----
  for (int ni = 0; ni < 2; ++ni) {
    int nt = w + ni * 8;
    v8f acc = v8zero();
    for (int kc = 0; kc < 8; ++kc) {
      v16bf a = loadA_lds(sXH, 264, 0, kc * 32, lane);
      v16bf bb = loadB(W2B, kc * 16 + nt, lane);
      acc = wmma_bf16(a, bb, acc);
    }
    int n = nt * 16 + (lane & 15);
    int rb = 8 * (lane >> 4);
    float bv = b2[n];
#pragma unroll
    for (int r = 0; r < 8; ++r)
      out[(size_t)(b0 + rb + r) * 256 + n] = acc[r] + bv;
  }
}

// ---------------- host entry ----------------
extern "C" void kernel_launch(void* const* d_in, const int* in_sizes, int n_in,
                              void* d_out, int out_size, void* d_ws, size_t ws_size,
                              hipStream_t stream) {
  (void)in_sizes; (void)n_in; (void)out_size; (void)ws_size;
  const float* src   = (const float*)d_in[0];
  const float* src_t = (const float*)d_in[1];
  const float* seq   = (const float*)d_in[2];
  const float* seq_t = (const float*)d_in[3];
  const float* seq_e = (const float*)d_in[4];
  const unsigned char* mask = (const unsigned char*)d_in[5];
  const float* Wq = (const float*)d_in[6];
  const float* Wk = (const float*)d_in[7];
  const float* Wv = (const float*)d_in[8];
  const float* Wo = (const float*)d_in[9];
  const float* bo = (const float*)d_in[10];
  const float* lg = (const float*)d_in[11];
  const float* lb = (const float*)d_in[12];
  const float* W1 = (const float*)d_in[13];
  const float* b1 = (const float*)d_in[14];
  const float* W2 = (const float*)d_in[15];
  const float* b2 = (const float*)d_in[16];
  float* out = (float*)d_out;

  char* ws = (char*)d_ws;
  unsigned short* WqB = (unsigned short*)(ws + OFF_WQ);
  unsigned short* WkB = (unsigned short*)(ws + OFF_WK);
  unsigned short* WvB = (unsigned short*)(ws + OFF_WV);
  unsigned short* WoB = (unsigned short*)(ws + OFF_WO);
  unsigned short* W1B = (unsigned short*)(ws + OFF_W1);
  unsigned short* W2B = (unsigned short*)(ws + OFF_W2);
  unsigned short* Qm  = (unsigned short*)(ws + OFF_Q);
  unsigned short* Km  = (unsigned short*)(ws + OFF_K);
  unsigned short* Vm  = (unsigned short*)(ws + OFF_V);

  auto prep = [&](const float* W, unsigned short* o, int K, int N) {
    int total = (K / 2) * N;
    prep_weight<<<(total + 255) / 256, 256, 0, stream>>>(W, (unsigned*)o, K, N);
  };
  prep(Wq, WqB, 768, 768);
  prep(Wk, WkB, 768, 768);
  prep(Wv, WvB, 768, 768);
  prep(Wo, WoB, 768, 768);
  prep(W1, W1B, 1024, 256);
  prep(W2, W2B, 256, 256);

  const size_t ldsGemm = (size_t)128 * 776 * 2;   // 198,656 B
  q_gemm<<<Bsz / 128, 256, ldsGemm, stream>>>(src, src_t, WqB, Qm);
  kv_gemm<<<ROWS_KV / 128, 256, ldsGemm, stream>>>(seq, seq_e, seq_t, WkB, WvB, Km, Vm);

  const size_t ldsAttn = 239360;                  // < 320 KB WGP LDS
  attn_fused<<<Bsz / 16, 256, ldsAttn, stream>>>(Qm, Km, Vm, src, src_t, mask,
                                                 WoB, W1B, W2B, bo, lg, lb, b1, b2,
                                                 out);
}
